// Model_57621281243601
// MI455X (gfx1250) — matrix operand order repaired and checked by reading, not compiled, not run on hardware
//
#include <hip/hip_runtime.h>
#include <hip/hip_bf16.h>
#include <cmath>
#include <cstddef>

// ---------------------------------------------------------------------------
// MACE-style equivariant GNN forward for MI455X (gfx1250, wave32, WMMA).
//
// Roofline: per layer, dense edge matmuls ~22 GFLOP (bf16 WMMA, trivial vs
// WMMA peak) while segment_sum scatters ~712 MB of f32 atomics (the real
// limiter at 23.3 TB/s-class memory). So: WMMA for all edge GEMMs, global
// f32 atomics for the scatter, plain VALU for the small node einsums.
// Fast v_rcp_f32 replaces IEEE divides so VALU stays free for address math.
//
// Workspace layout (floats), N = #nodes:
//   set stride SET = N*1112   (F0:N*128, F1:N*288, F2:N*360, F3:N*336)
//   [0,SET)      feats set A
//   [SET,2SET)   feats set B
//   [2SET,3SET)  edge aggregation (zeroed per layer)
//   then bf16-packed weight fragments (~140 KB).
// Requires ws_size >= 3*SET*4 + 256 KB  (~134 MB for N=10000).
// ---------------------------------------------------------------------------

typedef __attribute__((ext_vector_type(16))) __bf16 v16bf;
typedef __attribute__((ext_vector_type(8)))  float  v8f;

__device__ __forceinline__ float fast_rcp(float x) { return __builtin_amdgcn_rcpf(x); }
__device__ __forceinline__ float silu_f(float x) {
    return x * fast_rcp(1.f + __expf(-x));
}

// ---------------------------------------------------------------------------
// Repack f32 weight matrix W (Ksrc x Ncols, row-major) into per-lane bf16
// B-fragment layout for v_wmma_f32_16x16x32_bf16:
//   dst[nt][kt][lane][i] = W[kt*32 + i + 8*(i>=8) + 8*(lane>=16)][nt*16 + (lane&15)]
// ---------------------------------------------------------------------------
__global__ void repack_kernel(const float* __restrict__ W, unsigned short* __restrict__ dst,
                              int Ksrc, int Ncols, int NT, int KT) {
    int idx = blockIdx.x * blockDim.x + threadIdx.x;
    int total = NT * KT * 32 * 16;
    if (idx >= total) return;
    int i    = idx & 15;
    int lane = (idx >> 4) & 31;
    int kt   = (idx >> 9) % KT;
    int nt   = idx / (512 * KT);
    int K    = kt * 32 + i + ((i < 8) ? 0 : 8) + ((lane >= 16) ? 8 : 0);
    int col  = nt * 16 + (lane & 15);
    float v  = (K < Ksrc && col < Ncols) ? W[(size_t)K * Ncols + col] : 0.f;
    __bf16 b = (__bf16)v;
    dst[idx] = __builtin_bit_cast(unsigned short, b);
}

__global__ void zero_kernel(float* p, size_t n) {
    size_t i = (size_t)blockIdx.x * blockDim.x + threadIdx.x;
    if (i < n) p[i] = 0.f;
}

__global__ void embed_kernel(const int* __restrict__ species, const float* __restrict__ emb,
                             float* __restrict__ f0, int N) {
    int i = blockIdx.x * blockDim.x + threadIdx.x;
    if (i < N * 32) {
        int n = i / 32, c = i % 32;
        f0[i] = emb[species[n] * 32 + c];
    }
}

// ---------------------------------------------------------------------------
// Edge kernel: 4 waves per block, one 16-edge tile per wave.
//   hidden = silu(bessel(16x8) @ W1(8x64))          -> 4 WMMAs
//   w      = hidden(16x64) @ W2(64x344)             -> 2 WMMAs per 16-col tile
//   s      = feat0[sender](16xINMUL) @ Wp(INMULx344)-> INMUL/32 WMMAs per tile
//   scatter (w*s)[e,ch] * Y_l[e,k] / sqrt(20) into agg_l[receiver] (atomics)
// ---------------------------------------------------------------------------
template <int INMUL>
__global__ void __launch_bounds__(128)
edge_kernel(const float* __restrict__ pos, const int* __restrict__ senders,
            const int* __restrict__ receivers, const float* __restrict__ feat0,
            const unsigned short* __restrict__ w1pack, const unsigned short* __restrict__ w2pack,
            const unsigned short* __restrict__ wppack,
            float* __restrict__ agg0, float* __restrict__ agg1,
            float* __restrict__ agg2, float* __restrict__ agg3, int E) {
    constexpr int KT = INMUL / 32;

    __shared__ float sY[4][16][16];
    __shared__ int   sRecv[4][16];
    __shared__ float sHid[4][16][64];

    const int wid  = threadIdx.x >> 5;
    const int lane = threadIdx.x & 31;
    const int half = lane >> 4;
    const int l15  = lane & 15;
    const int edgeBase = (blockIdx.x * 4 + wid) * 16;
    const int e = edgeBase + l15;
    const bool valid = (e < E);

    // --- per-edge scalar setup: r, u, bessel, spherical harmonics -----------
    // Only lanes 0..15 consume this (A-fragment K<8 slots + LDS Y/recv table),
    // so predicate the transcendental work on half==0. Sender indices are
    // needed by both halves (A-fragment row gather).
    float rb[8];
    int sendIdx = valid ? senders[e] : 0;
    {
        float Y[16];
#pragma unroll
        for (int i = 0; i < 8; i++) rb[i] = 0.f;
#pragma unroll
        for (int i = 0; i < 16; i++) Y[i] = 0.f;
        int recv = -1;
        if (valid && half == 0) {
            int rcv = receivers[e];
            recv = rcv;
            int snd = sendIdx;
            float vx = (pos[rcv * 3 + 0] - pos[snd * 3 + 0]) * 0.5f;   // /CUTOFF
            float vy = (pos[rcv * 3 + 1] - pos[snd * 3 + 1]) * 0.5f;
            float vz = (pos[rcv * 3 + 2] - pos[snd * 3 + 2]) * 0.5f;
            float r   = sqrtf(vx * vx + vy * vy + vz * vz);
            float inv = fast_rcp(r + 1e-9f);
            float x = vx * inv, y = vy * inv, z = vz * inv;
            float env = (r < 1.f) ? (1.f - 3.f * r * r + 2.f * r * r * r) : 0.f;
            float ir  = env * fast_rcp(r + 1e-6f);
#pragma unroll
            for (int n = 0; n < 8; n++)
                rb[n] = __sinf(3.14159265358979323f * (float)(n + 1) * r) * ir;
            const float s3  = 1.7320508075688772f;   // sqrt(3)
            const float c15 = 3.8729833462074170f;   // sqrt(15)
            const float c5  = 2.2360679774997896f;   // sqrt(5)
            const float a3  = 2.0916500663351889f;   // sqrt(35/8)
            const float b3  = 10.246950765959598f;   // sqrt(105)
            const float c3  = 1.6201851746019651f;   // sqrt(21/8)
            const float d3  = 1.3228756555322954f;   // sqrt(7/4)
            const float e3  = 5.1234753829797990f;   // sqrt(105/4)
            Y[0] = 1.f;
            Y[1] = s3 * x; Y[2] = s3 * y; Y[3] = s3 * z;
            Y[4] = c15 * x * y; Y[5] = c15 * y * z;
            Y[6] = 0.5f * c5 * (3.f * z * z - 1.f);
            Y[7] = c15 * x * z; Y[8] = 0.5f * c15 * (x * x - y * y);
            Y[9]  = a3 * y * (3.f * x * x - y * y);
            Y[10] = b3 * x * y * z;
            Y[11] = c3 * y * (5.f * z * z - 1.f);
            Y[12] = d3 * z * (5.f * z * z - 3.f);
            Y[13] = c3 * x * (5.f * z * z - 1.f);
            Y[14] = e3 * z * (x * x - y * y);
            Y[15] = a3 * x * (x * x - 3.f * y * y);
        }
        if (half == 0) {
#pragma unroll
            for (int i = 0; i < 16; i++) sY[wid][l15][i] = Y[i];
            sRecv[wid][l15] = recv;
        }
    }

    // --- A fragment for bessel (16x32, K<8 live). A layout:
    //     K(i) = i + 8*(i>=8) + 8*(lane>=16)  -> only lane<16,i<8 are K<8.
    v16bf ab;
#pragma unroll
    for (int i = 0; i < 16; i++) {
        float v = (half == 0 && i < 8) ? rb[i] : 0.f;
        ab[i] = (__bf16)v;
    }

    const v16bf* W1F = reinterpret_cast<const v16bf*>(w1pack);
    const v16bf* W2F = reinterpret_cast<const v16bf*>(w2pack);
    const v16bf* WPF = reinterpret_cast<const v16bf*>(wppack);

    // Warm the packed-weight stream (global_prefetch_b8).
    __builtin_prefetch(&W2F[lane], 0, 0);
    __builtin_prefetch(&WPF[lane], 0, 0);

    // --- hidden = silu(bessel @ W1) : 16x64 --------------------------------
    v8f hid[4];
#pragma unroll
    for (int nt = 0; nt < 4; nt++) {
        v8f c = {};
        v16bf bw = W1F[nt * 32 + lane];
        c = __builtin_amdgcn_wmma_f32_16x16x32_bf16(false, ab, false, bw,
                                                    (short)0, c, false, false);
        hid[nt] = c;
    }
    __syncthreads();
    // D layout: element v of VGPR block -> (M = v + 8*half, N = l15)
#pragma unroll
    for (int nt = 0; nt < 4; nt++) {
#pragma unroll
        for (int v = 0; v < 8; v++) {
            int M = v + half * 8;
            sHid[wid][M][nt * 16 + l15] = silu_f(hid[nt][v]);
        }
    }
    __syncthreads();

    // --- A fragments of hidden (K=64 -> 2 k-tiles) -------------------------
    v16bf ah[2];
#pragma unroll
    for (int kt = 0; kt < 2; kt++) {
#pragma unroll
        for (int i = 0; i < 16; i++) {
            int K = kt * 32 + i + ((i < 8) ? 0 : 8) + half * 8;
            ah[kt][i] = (__bf16)sHid[wid][l15][K];
        }
    }

    // --- A fragments of gathered sender features (K=INMUL) -----------------
    v16bf af[KT];
    const float* frow = feat0 + (size_t)sendIdx * INMUL;
#pragma unroll
    for (int kt = 0; kt < KT; kt++) {
#pragma unroll
        for (int i = 0; i < 16; i++) {
            int K = kt * 32 + i + ((i < 8) ? 0 : 8) + half * 8;
            float v = valid ? frow[K] : 0.f;
            af[kt][i] = (__bf16)v;
        }
    }

    const float invAvg = 0.22360679774997896f;  // 1/sqrt(20)

    // --- channel tiles: 344 cols -> 22 tiles of 16 -------------------------
    for (int nt = 0; nt < 22; nt++) {
        // Prefetch next tile's packed B fragments ahead of the WMMAs.
        if (nt + 1 < 22) {
            __builtin_prefetch(&W2F[((nt + 1) * 2) * 32 + lane], 0, 0);
            __builtin_prefetch(&WPF[((nt + 1) * KT) * 32 + lane], 0, 0);
        }
        v8f wacc = {};
        v8f sacc = {};
#pragma unroll
        for (int kt = 0; kt < 2; kt++) {
            v16bf bw = W2F[(nt * 2 + kt) * 32 + lane];
            wacc = __builtin_amdgcn_wmma_f32_16x16x32_bf16(false, ah[kt], false, bw,
                                                           (short)0, wacc, false, false);
        }
#pragma unroll
        for (int kt = 0; kt < KT; kt++) {
            v16bf bw = WPF[(nt * KT + kt) * 32 + lane];
            sacc = __builtin_amdgcn_wmma_f32_16x16x32_bf16(false, af[kt], false, bw,
                                                           (short)0, sacc, false, false);
        }
        int ch = nt * 16 + l15;
        if (ch < 344) {
            int l, c0;
            if (ch < 128)      { l = 0; c0 = ch; }
            else if (ch < 224) { l = 1; c0 = ch - 128; }
            else if (ch < 296) { l = 2; c0 = ch - 224; }
            else               { l = 3; c0 = ch - 296; }
            const int comps = 2 * l + 1;
            const int ml = (l == 0) ? 128 : (l == 1) ? 96 : (l == 2) ? 72 : 48;
            const int yb = (l == 0) ? 0 : (l == 1) ? 1 : (l == 2) ? 4 : 9;
            float* aggp = (l == 0) ? agg0 : (l == 1) ? agg1 : (l == 2) ? agg2 : agg3;
#pragma unroll
            for (int v = 0; v < 8; v++) {
                int M = v + half * 8;
                int rcv = sRecv[wid][M];
                if (rcv >= 0) {
                    float val = wacc[v] * sacc[v] * invAvg;
                    float* dst = aggp + ((size_t)rcv * ml + c0) * comps;
                    for (int k = 0; k < comps; k++)
                        atomicAdd(dst + k, val * sY[wid][M][yb + k]);
                }
            }
        }
        __syncthreads();  // keep waves together; sY/sHid stay live
    }
}

// ---------------------------------------------------------------------------
// Node update: h[n,d,k] = sum_c x[n,c,k] * S[c,d]   (x = concat(prev, agg))
// ---------------------------------------------------------------------------
__global__ void node_kernel(const float* __restrict__ prev, const float* __restrict__ agg,
                            const float* __restrict__ S, float* __restrict__ out,
                            int N, int cin_prev, int m, int comps, int dosilu) {
    int id = blockIdx.x * blockDim.x + threadIdx.x;
    if (id >= N * m) return;
    int n = id / m, d = id % m;
    for (int k = 0; k < comps; k++) {
        float acc = 0.f;
        for (int c = 0; c < cin_prev; c++)
            acc += prev[((size_t)n * cin_prev + c) * comps + k] * S[c * m + d];
        for (int c = 0; c < m; c++)
            acc += agg[((size_t)n * m + c) * comps + k] * S[(cin_prev + c) * m + d];
        if (dosilu) acc = silu_f(acc);
        out[((size_t)n * m + d) * comps + k] = acc;
    }
}

// ---------------------------------------------------------------------------
// Final reduction: means, linear heads, contraction with Q tensors.
// ---------------------------------------------------------------------------
struct QConst {
    float q1[81];   // (3,3,3,3) row-major
    float q3[189];  // (3,3,3,7) row-major
};

__global__ void final_kernel(const float* __restrict__ f1, const float* __restrict__ f3,
                             const float* __restrict__ wlin1, const float* __restrict__ wlin3,
                             float* __restrict__ out, int N, float invN, QConst qc) {
    __shared__ float s1[288];
    __shared__ float s3buf[336];
    __shared__ float v[10];
    int t = threadIdx.x;
    for (int e2 = t; e2 < 288; e2 += 256) {
        float acc = 0.f;
        for (int n = 0; n < N; n++) acc += f1[(size_t)n * 288 + e2];
        s1[e2] = acc * invN;
    }
    for (int e2 = t; e2 < 336; e2 += 256) {
        float acc = 0.f;
        for (int n = 0; n < N; n++) acc += f3[(size_t)n * 336 + e2];
        s3buf[e2] = acc * invN;
    }
    __syncthreads();
    if (t < 3) {
        float a = 0.f;
        for (int c = 0; c < 96; c++) a += s1[c * 3 + t] * wlin1[c];
        v[t] = a;
    } else if (t < 10) {
        int m = t - 3;
        float a = 0.f;
        for (int c = 0; c < 48; c++) a += s3buf[c * 7 + m] * wlin3[c];
        v[t] = a;
    }
    __syncthreads();
    if (t < 27) {
        float a = 0.f;
        for (int s = 0; s < 3; s++) a += qc.q1[t * 3 + s] * v[s];
        for (int m = 0; m < 7; m++) a += qc.q3[t * 7 + m] * v[3 + m];
        out[t] = a;
    }
}

// ---------------------------------------------------------------------------
// CPU-side construction of Q1/Q3 (deterministic each call). The reference
// uses QR+SVD; we build the same subspaces with Gram-Schmidt (same span,
// different orthonormal basis for Q3 — no canonical SVD basis exists anyway).
// ---------------------------------------------------------------------------
static void computeQ(QConst& qc) {
    auto D = [](int a, int b) { return a == b ? 1.0 : 0.0; };
    double M[27][3];
    for (int i = 0; i < 3; i++)
        for (int j = 0; j < 3; j++)
            for (int k = 0; k < 3; k++) {
                int r = (i * 3 + j) * 3 + k;
                for (int s = 0; s < 3; s++)
                    M[r][s] = D(i, j) * D(k, s) + D(j, k) * D(i, s) + D(k, i) * D(j, s);
            }
    for (int s = 0; s < 3; s++) {
        double n = 0;
        for (int r = 0; r < 27; r++) n += M[r][s] * M[r][s];
        n = std::sqrt(n);
        for (int r = 0; r < 27; r++) M[r][s] /= n;
    }
    for (int r = 0; r < 27; r++)
        for (int s = 0; s < 3; s++) qc.q1[r * 3 + s] = (float)M[r][s];

    double Qm[27][3];
    int nq = 0;
    for (int s = 0; s < 3; s++) {
        double vv[27];
        for (int r = 0; r < 27; r++) vv[r] = M[r][s];
        for (int t = 0; t < nq; t++) {
            double dp = 0;
            for (int r = 0; r < 27; r++) dp += Qm[r][t] * vv[r];
            for (int r = 0; r < 27; r++) vv[r] -= dp * Qm[r][t];
        }
        double n = 0;
        for (int r = 0; r < 27; r++) n += vv[r] * vv[r];
        n = std::sqrt(n);
        if (n > 1e-12) {
            for (int r = 0; r < 27; r++) Qm[r][nq] = vv[r] / n;
            nq++;
        }
    }
    double Bm[27][10];
    int col = 0;
    for (int i = 0; i < 3; i++)
        for (int j = i; j < 3; j++)
            for (int k = j; k < 3; k++) {
                double tt[27];
                for (int r = 0; r < 27; r++) tt[r] = 0.0;
                int perms[6][3] = {{i, j, k}, {i, k, j}, {j, i, k}, {j, k, i}, {k, i, j}, {k, j, i}};
                for (int p = 0; p < 6; p++) {
                    int r = (perms[p][0] * 3 + perms[p][1]) * 3 + perms[p][2];
                    tt[r] += 1.0;
                }
                for (int r = 0; r < 27; r++) Bm[r][col] = tt[r];
                col++;
            }
    for (int c = 0; c < 10; c++)
        for (int t = 0; t < nq; t++) {
            double dp = 0;
            for (int r = 0; r < 27; r++) dp += Qm[r][t] * Bm[r][c];
            for (int r = 0; r < 27; r++) Bm[r][c] -= dp * Qm[r][t];
        }
    double U[27][7];
    int nu = 0;
    for (int c = 0; c < 10 && nu < 7; c++) {
        double vv[27];
        for (int r = 0; r < 27; r++) vv[r] = Bm[r][c];
        for (int t = 0; t < nu; t++) {
            double dp = 0;
            for (int r = 0; r < 27; r++) dp += U[r][t] * vv[r];
            for (int r = 0; r < 27; r++) vv[r] -= dp * U[r][t];
        }
        double n = 0;
        for (int r = 0; r < 27; r++) n += vv[r] * vv[r];
        n = std::sqrt(n);
        if (n > 1e-6) {
            for (int r = 0; r < 27; r++) U[r][nu] = vv[r] / n;
            nu++;
        }
    }
    for (int r = 0; r < 27; r++)
        for (int m = 0; m < 7; m++) qc.q3[r * 7 + m] = (float)(m < nu ? U[r][m] : 0.0);
}

// ---------------------------------------------------------------------------
extern "C" void kernel_launch(void* const* d_in, const int* in_sizes, int n_in,
                              void* d_out, int out_size, void* d_ws, size_t ws_size,
                              hipStream_t stream) {
    const float* positions = (const float*)d_in[0];
    const int*   species   = (const int*)d_in[1];
    const int*   senders   = (const int*)d_in[2];
    const int*   receivers = (const int*)d_in[3];
    const float* emb       = (const float*)d_in[4];
    const float* wlin1     = (const float*)d_in[26];
    const float* wlin3     = (const float*)d_in[27];
    float* out = (float*)d_out;

    const int N = in_sizes[0] / 3;
    const int E = in_sizes[2];

    float* ws = (float*)d_ws;
    const size_t SET = (size_t)N * 1112;
    float* FA = ws;
    float* FB = ws + SET;
    float* AG = ws + 2 * SET;
    unsigned short* pack = (unsigned short*)(ws + 3 * SET);
    unsigned short* w1p = pack;                  // 4*1*512   = 2048
    unsigned short* w2p = w1p + 2048;            // 22*2*512  = 22528
    unsigned short* wpp = w2p + 22528;           // 22*4*512  = 45056 max

    const size_t o1 = (size_t)N * 128;
    const size_t o2 = (size_t)N * 416;
    const size_t o3 = (size_t)N * 776;

    QConst qc;
    computeQ(qc);

    embed_kernel<<<(N * 32 + 255) / 256, 256, 0, stream>>>(species, emb, FA, N);

    const int   inmul[3] = {32, 128, 128};
    float*      fin[3]   = {FA, FB, FA};   // edge-kernel feat0 input
    float*      fprev[3] = {FA, FB, FA};   // node-kernel prev feats
    float*      fout[3]  = {FB, FA, FB};   // node-kernel output set
    const int   MULS[4]  = {128, 96, 72, 48};

    for (int li = 0; li < 3; li++) {
        const float* lW1 = (const float*)d_in[5 + li * 7 + 0];
        const float* lW2 = (const float*)d_in[5 + li * 7 + 1];
        const float* lWp = (const float*)d_in[5 + li * 7 + 2];
        const float* Sl[4] = {(const float*)d_in[5 + li * 7 + 3], (const float*)d_in[5 + li * 7 + 4],
                              (const float*)d_in[5 + li * 7 + 5], (const float*)d_in[5 + li * 7 + 6]};

        zero_kernel<<<(unsigned)((SET + 255) / 256), 256, 0, stream>>>(AG, SET);

        repack_kernel<<<(4 * 1 * 512 + 255) / 256, 256, 0, stream>>>(lW1, w1p, 8, 64, 4, 1);
        repack_kernel<<<(22 * 2 * 512 + 255) / 256, 256, 0, stream>>>(lW2, w2p, 64, 344, 22, 2);
        {
            int kt = inmul[li] / 32;
            repack_kernel<<<(22 * kt * 512 + 255) / 256, 256, 0, stream>>>(lWp, wpp, inmul[li], 344, 22, kt);
        }

        int eblocks = (E + 63) / 64;
        if (inmul[li] == 32)
            edge_kernel<32><<<eblocks, 128, 0, stream>>>(positions, senders, receivers, fin[li],
                                                         w1p, w2p, wpp,
                                                         AG, AG + o1, AG + o2, AG + o3, E);
        else
            edge_kernel<128><<<eblocks, 128, 0, stream>>>(positions, senders, receivers, fin[li],
                                                          w1p, w2p, wpp,
                                                          AG, AG + o1, AG + o2, AG + o3, E);

        for (int l = 0; l < 4; l++) {
            int comps = 2 * l + 1;
            int m = MULS[l];
            int cinp = (li == 0) ? (l == 0 ? 32 : 0) : m;
            size_t off = (l == 0) ? 0 : (l == 1) ? o1 : (l == 2) ? o2 : o3;
            const float* prevp = (li == 0) ? (l == 0 ? fprev[li] : nullptr) : (fprev[li] + off);
            float* aggp = AG + off;
            float* outp = fout[li] + off;
            int threads = N * m;
            node_kernel<<<(threads + 255) / 256, 256, 0, stream>>>(prevp, aggp, Sl[l], outp,
                                                                   N, cinp, m, comps, l == 0 ? 1 : 0);
        }
    }

    final_kernel<<<1, 256, 0, stream>>>(FB + o1, FB + o3, wlin1, wlin3, out, N,
                                        1.0f / (float)N, qc);
}
